// RNN_31155692765764
// MI455X (gfx1250) — matrix operand order repaired and checked by reading, not compiled, not run on hardware
//
#include <hip/hip_runtime.h>
#include <stdint.h>

typedef __attribute__((ext_vector_type(16))) _Float16 v16h;
typedef __attribute__((ext_vector_type(8)))  _Float16 v8h;
typedef __attribute__((ext_vector_type(8)))  float    v8f;

#define L_SEQ 100000
#define H     256
#define TC    64
#define NBUF  2
#define NCH   ((L_SEQ + TC - 1) / TC)

// workspace layout (bytes):
//   [0, 64)        : int flags[16]   0=h1_count 1=h1_consumed 2=z_count 3=z_consumed
//   [256, +64KB)   : h1 ring, NBUF*TC*H fp16
//   [+.., +128KB)  : z  ring, NBUF*TC*H f32
#define H1_OFF 256
#define Z_OFF  (256 + (size_t)NBUF * TC * H * 2)

static __device__ inline v16h cat8(v8h lo, v8h hi) {
    v16h r;
#pragma unroll
    for (int i = 0; i < 8; ++i) { r[i] = lo[i]; r[8 + i] = hi[i]; }
    return r;
}

// Build an A-matrix (16x32 f16) fragment from a row-major fp32 weight matrix.
// Lane layout (contiguous K, this code's order, not the ISA table): lanes 0-15 -> M rows, g = lane>>4 selects K halves:
//   elems 0..15 = W[row, k0 + 16g .. +15]
static __device__ inline v16h load_afrag(const float* W, int row, int k0, int g) {
    v16h r;
    const float* p = W + (size_t)row * H + k0;
#pragma unroll
    for (int j = 0; j < 8; ++j) r[j]     = (_Float16)p[16 * g + j];
#pragma unroll
    for (int j = 0; j < 8; ++j) r[8 + j] = (_Float16)p[16 * g + 8 + j];
    return r;
}

static __device__ inline v8f wmma_f16(v16h a, v16h b, v8f c) {
    return __builtin_amdgcn_wmma_f32_16x16x32_f16(
        /*neg_a=*/false, a, /*neg_b=*/false, b,
        /*c_mod=*/(short)0, c, /*reuse_a=*/false, /*reuse_b=*/false);
}

static __device__ inline void wait_ge(int* f, int target) {
    while (__hip_atomic_load(f, __ATOMIC_ACQUIRE, __HIP_MEMORY_SCOPE_AGENT) < target)
        __builtin_amdgcn_s_sleep(2);
}
static __device__ inline void post(int* f, int v) {
    __hip_atomic_store(f, v, __ATOMIC_RELEASE, __HIP_MEMORY_SCOPE_AGENT);
}

__global__ void __launch_bounds__(256)
rnn_init_flags(int* flags) {
    if (threadIdx.x < 16) flags[threadIdx.x] = 0;
}

__global__ void __launch_bounds__(256)
rnn_pipeline(const float* __restrict__ x,     const float* __restrict__ h0,
             const float* __restrict__ W_ih0, const float* __restrict__ W_hh0,
             const float* __restrict__ b_ih0, const float* __restrict__ b_hh0,
             const float* __restrict__ W_ih1, const float* __restrict__ W_hh1,
             const float* __restrict__ b_ih1, const float* __restrict__ b_hh1,
             const float* __restrict__ W_out, const float* __restrict__ b_out,
             float* __restrict__ out, int* flags, void* h1ring_v, float* zring) {
    _Float16* h1ring = (_Float16*)h1ring_v;

    __shared__ __align__(32) _Float16 hlds[H]; // current hidden state (fp16)
    __shared__ float ylds[H];                  // matvec partial results
    __shared__ float red[32];                  // output reduction (8 waves x 4)

    const int tid  = threadIdx.x;
    const int lane = tid & 31;
    const int w    = tid >> 5;      // wave id (0..7)
    const int g    = lane >> 4;     // lane half-group
    const int m    = lane & 15;

    if (blockIdx.x == 0) {
        // ---- Stage 0: serial scan of layer 0 (h1_t = tanh(xW + W_hh0 h1)) ----
        const float wi0 = W_ih0[tid * 3 + 0], wi1 = W_ih0[tid * 3 + 1], wi2 = W_ih0[tid * 3 + 2];
        const float b0  = b_ih0[tid] + b_hh0[tid];
        v16h Wf[2][8]; // persistent W_hh0 A-fragments: 2 N-tiles x 8 K-blocks
#pragma unroll
        for (int tt = 0; tt < 2; ++tt) {
            const int row = (2 * w + tt) * 16 + m;
#pragma unroll
            for (int kb = 0; kb < 8; ++kb) Wf[tt][kb] = load_afrag(W_hh0, row, kb * 32, g);
        }
        hlds[tid] = (_Float16)h0[tid];
        __syncthreads();

        for (int c = 0; c < NCH; ++c) {
            if (tid == 0) wait_ge(&flags[1], c - NBUF + 1); // ring back-pressure
            __syncthreads();
            const int slot = c & 1;
            const int tend = (L_SEQ - c * TC < TC) ? (L_SEQ - c * TC) : TC;
            for (int i = 0; i < tend; ++i) {
                const int t = c * TC + i;
                const float xw = b0 + wi0 * x[3 * t] + wi1 * x[3 * t + 1] + wi2 * x[3 * t + 2];
                v8f acc0 = {}, acc1 = {};
#pragma unroll
                for (int kb = 0; kb < 8; ++kb) {
                    // B = h replicated across N: 16 contiguous f16 per lane-group
                    v16h bf = *(const v16h*)&hlds[kb * 32 + 16 * g];
                    acc0 = wmma_f16(Wf[0][kb], bf, acc0);
                    acc1 = wmma_f16(Wf[1][kb], bf, acc1);
                }
                if (m == 0) { // D replicated over N: lanes 0 and 16 carry M=r / M=8+r
                    int b0i = (2 * w + 0) * 16 + g * 8;
                    int b1i = (2 * w + 1) * 16 + g * 8;
#pragma unroll
                    for (int r = 0; r < 8; ++r) { ylds[b0i + r] = acc0[r]; ylds[b1i + r] = acc1[r]; }
                }
                __syncthreads();
                const float hn = tanhf(xw + ylds[tid]);
                const _Float16 hb = (_Float16)hn;
                hlds[tid] = hb;
                h1ring[(size_t)(slot * TC + i) * H + tid] = hb;
                __syncthreads();
            }
            __threadfence();
            __syncthreads();
            if (tid == 0) post(&flags[0], c + 1);
        }
    } else if (blockIdx.x == 1) {
        // ---- Stage 1: parallel GEMM z = h1_chunk @ W_ih1^T (TCx256 @ 256x256) ----
        v16h Bf[2][8]; // persistent W_ih1 B-fragments: 2 N-tiles x 8 K-blocks
#pragma unroll
        for (int tt = 0; tt < 2; ++tt) {
            const int nrow = (2 * w + tt) * 16 + m;
#pragma unroll
            for (int kb = 0; kb < 8; ++kb) {
                v16h r;
                const float* p = W_ih1 + (size_t)nrow * H + kb * 32 + 16 * g;
#pragma unroll
                for (int j = 0; j < 16; ++j) r[j] = (_Float16)p[j];
                Bf[tt][kb] = r;
            }
        }
        for (int c = 0; c < NCH; ++c) {
            if (tid == 0) { wait_ge(&flags[0], c + 1); wait_ge(&flags[3], c - NBUF + 1); }
            __syncthreads();
            const int slot = c & 1;
#pragma unroll 1
            for (int mt = 0; mt < TC / 16; ++mt) {
                v8f acc0 = {}, acc1 = {};
                const _Float16* arow = h1ring + (size_t)(slot * TC + mt * 16 + m) * H;
#pragma unroll
                for (int kb = 0; kb < 8; ++kb) {
                    v8h lo = *(const v8h*)(arow + kb * 32 + 16 * g);
                    v8h hi = *(const v8h*)(arow + kb * 32 + 16 * g + 8);
                    v16h af = cat8(lo, hi);
                    acc0 = wmma_f16(af, Bf[0][kb], acc0);
                    acc1 = wmma_f16(af, Bf[1][kb], acc1);
                }
                const int mrow0 = mt * 16 + g * 8;
                const int nc0 = (2 * w + 0) * 16 + m;
                const int nc1 = (2 * w + 1) * 16 + m;
#pragma unroll
                for (int r = 0; r < 8; ++r) {
                    zring[(size_t)(slot * TC + mrow0 + r) * H + nc0] = acc0[r];
                    zring[(size_t)(slot * TC + mrow0 + r) * H + nc1] = acc1[r];
                }
            }
            __threadfence();
            __syncthreads();
            if (tid == 0) { post(&flags[1], c + 1); post(&flags[2], c + 1); }
        }
    } else {
        // ---- Stage 2: serial scan of layer 1 + fused output projection ----
        const float b1 = b_ih1[tid] + b_hh1[tid];
        const float wo0 = W_out[0 * H + tid], wo1 = W_out[1 * H + tid];
        const float wo2 = W_out[2 * H + tid], wo3 = W_out[3 * H + tid];
        v16h Wf[2][8];
#pragma unroll
        for (int tt = 0; tt < 2; ++tt) {
            const int row = (2 * w + tt) * 16 + m;
#pragma unroll
            for (int kb = 0; kb < 8; ++kb) Wf[tt][kb] = load_afrag(W_hh1, row, kb * 32, g);
        }
        hlds[tid] = (_Float16)h0[H + tid];
        __syncthreads();

        for (int c = 0; c < NCH; ++c) {
            if (tid == 0) wait_ge(&flags[2], c + 1);
            __syncthreads();
            const int slot = c & 1;
            const int tend = (L_SEQ - c * TC < TC) ? (L_SEQ - c * TC) : TC;
            for (int i = 0; i < tend; ++i) {
                const int t = c * TC + i;
                v8f acc0 = {}, acc1 = {};
#pragma unroll
                for (int kb = 0; kb < 8; ++kb) {
                    v16h bf = *(const v16h*)&hlds[kb * 32 + 16 * g];
                    acc0 = wmma_f16(Wf[0][kb], bf, acc0);
                    acc1 = wmma_f16(Wf[1][kb], bf, acc1);
                }
                if (m == 0) {
                    int b0i = (2 * w + 0) * 16 + g * 8;
                    int b1i = (2 * w + 1) * 16 + g * 8;
#pragma unroll
                    for (int r = 0; r < 8; ++r) { ylds[b0i + r] = acc0[r]; ylds[b1i + r] = acc1[r]; }
                }
                __syncthreads();
                const float zv = zring[(size_t)(slot * TC + i) * H + tid];
                const float hn = tanhf(zv + b1 + ylds[tid]);
                hlds[tid] = (_Float16)hn;
                // fused out_t = h2 @ W_out^T + b_out (wave shuffle reduce, then LDS)
                float c0 = hn * wo0, c1 = hn * wo1, c2 = hn * wo2, c3 = hn * wo3;
#pragma unroll
                for (int off = 16; off >= 1; off >>= 1) {
                    c0 += __shfl_xor(c0, off);
                    c1 += __shfl_xor(c1, off);
                    c2 += __shfl_xor(c2, off);
                    c3 += __shfl_xor(c3, off);
                }
                if (lane == 0) { red[w * 4 + 0] = c0; red[w * 4 + 1] = c1; red[w * 4 + 2] = c2; red[w * 4 + 3] = c3; }
                __syncthreads();
                if (tid < 4) {
                    float s = b_out[tid];
#pragma unroll
                    for (int ww = 0; ww < 8; ++ww) s += red[ww * 4 + tid];
                    out[(size_t)t * 4 + tid] = s;
                }
                __syncthreads();
            }
            __syncthreads();
            if (tid == 0) post(&flags[3], c + 1);
        }
    }
}

extern "C" void kernel_launch(void* const* d_in, const int* in_sizes, int n_in,
                              void* d_out, int out_size, void* d_ws, size_t ws_size,
                              hipStream_t stream) {
    (void)in_sizes; (void)n_in; (void)out_size; (void)ws_size;
    const float* x     = (const float*)d_in[0];
    const float* h0    = (const float*)d_in[1];
    const float* W_ih0 = (const float*)d_in[2];
    const float* W_hh0 = (const float*)d_in[3];
    const float* b_ih0 = (const float*)d_in[4];
    const float* b_hh0 = (const float*)d_in[5];
    const float* W_ih1 = (const float*)d_in[6];
    const float* W_hh1 = (const float*)d_in[7];
    const float* b_ih1 = (const float*)d_in[8];
    const float* b_hh1 = (const float*)d_in[9];
    const float* W_out = (const float*)d_in[10];
    const float* b_out = (const float*)d_in[11];
    float* out = (float*)d_out;

    char* ws     = (char*)d_ws;
    int*  flags  = (int*)ws;
    void* h1ring = (void*)(ws + H1_OFF);
    float* zring = (float*)(ws + Z_OFF);

    hipLaunchKernelGGL(rnn_init_flags, dim3(1), dim3(64), 0, stream, flags);
    hipLaunchKernelGGL(rnn_pipeline, dim3(3), dim3(256), 0, stream,
                       x, h0, W_ih0, W_hh0, b_ih0, b_hh0,
                       W_ih1, W_hh1, b_ih1, b_hh1, W_out, b_out,
                       out, flags, h1ring, zring);
}
